// MahalanobisLayer_34187939676803
// MI455X (gfx1250) — matrix operand order repaired and checked by reading, not compiled, not run on hardware
//
#include <hip/hip_runtime.h>

typedef float  v8f   __attribute__((ext_vector_type(8)));
typedef __bf16 v8bf  __attribute__((ext_vector_type(8)));
typedef __bf16 v16bf __attribute__((ext_vector_type(16)));

#define DIM      1024
#define ROW_PAD  8
#define LROW     (DIM + ROW_PAD)      // padded LDS row stride in elements
#define ROWS_PER_WG 32                // two 16-row M-tiles share each B fragment
#define NTILES   64                   // DIM/16 column tiles
#define KBLOCKS  32                   // DIM/32 k-steps
#define NT_PER_WAVE 8                 // 64 tiles / 8 waves

static __device__ __forceinline__ unsigned short f32_to_bf16(float f) {
    unsigned int u = __float_as_uint(f);
    unsigned int r = 0x7FFFu + ((u >> 16) & 1u);   // round to nearest even
    return (unsigned short)((u + r) >> 16);
}
static __device__ __forceinline__ float bf16_to_f32(unsigned short s) {
    return __uint_as_float(((unsigned int)s) << 16);
}

// ---------------------------------------------------------------------------
// Kernel 1: repack S_inv (f32, row-major [K][N]) into bf16 B-fragments.
// Destination layout: Bp[((ntile*KBLOCKS + kb)*32 + lane)*16 + e]
//   element e of lane l holds B[K][N] with
//   K = kb*32 + (l>>4)*16 + e,  N = ntile*16 + (l&15)
// Mapping is iterated over the SOURCE so the f32 reads are fully coalesced.
// ---------------------------------------------------------------------------
__global__ void __launch_bounds__(256)
mahal_pack_sinv_bf16(const float* __restrict__ S, unsigned short* __restrict__ Bp) {
    int idx = blockIdx.x * 256 + threadIdx.x;   // 0 .. 1048575, source order
    int K = idx >> 10;
    int N = idx & (DIM - 1);
    int kb    = K >> 5;
    int ko    = K & 31;
    int h     = ko >> 4;
    int e     = ko & 15;
    int ntile = N >> 4;
    int nl    = N & 15;
    int lane  = h * 16 + nl;
    Bp[(((size_t)ntile * KBLOCKS + kb) * 32 + lane) * 16 + e] = f32_to_bf16(S[idx]);
}

// ---------------------------------------------------------------------------
// Kernel 2: per 32-row tile, compute rowsum( (delta @ S_inv) ⊙ delta ).
// kb is the OUTER loop: A fragments loaded once per kb from LDS and reused
// across 8 n-tiles; 16 v8f accumulators (128 VGPRs) live across the loop.
// Each B fragment feeds two WMMAs (two M-tiles) -> half the L2 B traffic.
// ---------------------------------------------------------------------------
__global__ void __launch_bounds__(256)
mahal_wmma_kernel(const float* __restrict__ x,
                  const float* __restrict__ xfit,
                  const unsigned short* __restrict__ Bp,
                  float* __restrict__ out) {
    __shared__ unsigned short sd[ROWS_PER_WG * LROW];  // delta, bf16, padded rows
    __shared__ float sacc[ROWS_PER_WG];

    const int tile = blockIdx.x;           // 32-row tile index (0..511)
    const int tid  = threadIdx.x;

    if (tid < ROWS_PER_WG) sacc[tid] = 0.0f;

    // ---- stage delta = x - x_fit as bf16 into LDS (coalesced 8B/thread) ----
    const float* xb = x    + (size_t)tile * ROWS_PER_WG * DIM;
    const float* fb = xfit + (size_t)tile * ROWS_PER_WG * DIM;
#pragma unroll
    for (int j = 0; j < 64; ++j) {
        int p   = j * 256 + tid;           // pair index 0..16383
        int e0  = p * 2;
        int row = e0 >> 10;
        int col = e0 & (DIM - 1);
        float d0 = xb[e0]     - fb[e0];
        float d1 = xb[e0 + 1] - fb[e0 + 1];
        unsigned int packed = (unsigned int)f32_to_bf16(d0)
                            | ((unsigned int)f32_to_bf16(d1) << 16);
        *(unsigned int*)&sd[row * LROW + col] = packed;
    }
    __syncthreads();

    const int wave = tid >> 5;
    const int lane = tid & 31;
    const int m    = lane & 15;     // A row within M-tile / C column-in-tile
    const int h    = lane >> 4;     // lane half selector

    const unsigned short* arow0 = sd + m * LROW;            // M-tile 0
    const unsigned short* arow1 = sd + (16 + m) * LROW;     // M-tile 1

    // per-(wave,lane) base of the B fragment stream; fragment for (t, kb) is
    // at bwave + (t*KBLOCKS + kb)*512 elements (1024 bytes)
    const unsigned short* bwave =
        Bp + (((size_t)(wave * NT_PER_WAVE) * KBLOCKS) * 32 + lane) * 16;

    v8f acc0[NT_PER_WAVE];
    v8f acc1[NT_PER_WAVE];
#pragma unroll
    for (int t = 0; t < NT_PER_WAVE; ++t) { acc0[t] = (v8f){}; acc1[t] = (v8f){}; }

#pragma unroll 2
    for (int kb = 0; kb < KBLOCKS; ++kb) {
        // A fragments (16x32 bf16): elems 0..15 = K kb*32+h*16..
        const int k0 = kb * 32 + h * 16;
        v8bf lo0 = *(const v8bf*)(arow0 + k0);
        v8bf hi0 = *(const v8bf*)(arow0 + k0 + 8);
        v16bf a0 = __builtin_shufflevector(lo0, hi0,
                     0, 1, 2, 3, 4, 5, 6, 7, 8, 9, 10, 11, 12, 13, 14, 15);
        v8bf lo1 = *(const v8bf*)(arow1 + k0);
        v8bf hi1 = *(const v8bf*)(arow1 + k0 + 8);
        v16bf a1 = __builtin_shufflevector(lo1, hi1,
                     0, 1, 2, 3, 4, 5, 6, 7, 8, 9, 10, 11, 12, 13, 14, 15);

#pragma unroll
        for (int t = 0; t < NT_PER_WAVE; ++t) {
            // B fragment: 32 contiguous bytes per lane, reused for both M-tiles
            v16bf b = *(const v16bf*)(bwave + ((size_t)t * KBLOCKS + kb) * 512);
            acc0[t] = __builtin_amdgcn_wmma_f32_16x16x32_bf16(
                          false, a0, false, b, (short)0, acc0[t], false, false);
            acc1[t] = __builtin_amdgcn_wmma_f32_16x16x32_bf16(
                          false, a1, false, b, (short)0, acc1[t], false, false);
        }
    }

    // acc[t][r] = T[M = r + h*8 (+16)][N = (wave*8+t)*16 + m]; fold in delta,
    // then reduce per-row partials across lanes/waves via LDS float atomics.
    float psum0[8], psum1[8];
#pragma unroll
    for (int r = 0; r < 8; ++r) { psum0[r] = 0.0f; psum1[r] = 0.0f; }

#pragma unroll
    for (int t = 0; t < NT_PER_WAVE; ++t) {
        const int ncol = (wave * NT_PER_WAVE + t) * 16 + m;
#pragma unroll
        for (int r = 0; r < 8; ++r) {
            float d0 = bf16_to_f32(sd[(r + h * 8) * LROW + ncol]);
            float d1 = bf16_to_f32(sd[(16 + r + h * 8) * LROW + ncol]);
            psum0[r] += acc0[t][r] * d0;
            psum1[r] += acc1[t][r] * d1;
        }
    }

#pragma unroll
    for (int r = 0; r < 8; ++r) {
        atomicAdd(&sacc[r + h * 8],      psum0[r]);
        atomicAdd(&sacc[16 + r + h * 8], psum1[r]);
    }
    __syncthreads();

    if (tid < ROWS_PER_WG) out[(size_t)tile * ROWS_PER_WG + tid] = sacc[tid];
}

extern "C" void kernel_launch(void* const* d_in, const int* in_sizes, int n_in,
                              void* d_out, int out_size, void* d_ws, size_t ws_size,
                              hipStream_t stream) {
    (void)in_sizes; (void)n_in; (void)out_size; (void)ws_size;
    const float* x    = (const float*)d_in[0];
    const float* xfit = (const float*)d_in[1];
    const float* Sinv = (const float*)d_in[2];
    float* out = (float*)d_out;
    unsigned short* Bp = (unsigned short*)d_ws;   // 2 MB packed bf16 S_inv

    // 1024*1024 = 1,048,576 source elements
    mahal_pack_sinv_bf16<<<4096, 256, 0, stream>>>(Sinv, Bp);
    // 16384 rows / 32 rows-per-WG = 512 workgroups
    mahal_wmma_kernel<<<512, 256, 0, stream>>>(x, xfit, Bp, out);
}
